// DynamicParametersBehaviour_11501922418800
// MI455X (gfx1250) — matrix operand order repaired and checked by reading, not compiled, not run on hardware
//
#include <hip/hip_runtime.h>

typedef float        v8f   __attribute__((ext_vector_type(8)));
typedef __bf16       v16bf __attribute__((ext_vector_type(16)));
typedef __bf16       v8bf  __attribute__((ext_vector_type(8)));
typedef unsigned int v4u   __attribute__((ext_vector_type(4)));
typedef unsigned int v8u   __attribute__((ext_vector_type(8)));

#define HID   1024
#define DIN   2048
#define DOUT  2048
#define DEPTH 128
#define NB    16
#define MTOT  (NB * DEPTH)   // 2048 rows of logits / IV / OV
#define CLAMP_V 5.0f

struct U8x32 { v4u a, b; };
static __device__ __forceinline__ v16bf frag_ab(v4u a, v4u b) {
    U8x32 t; t.a = a; t.b = b;
    return __builtin_bit_cast(v16bf, t);
}
static __device__ __forceinline__ v16bf frag_c(v8u c) {
    return __builtin_bit_cast(v16bf, c);
}
static __device__ __forceinline__ v8f wmma_bf16(v16bf a, v16bf b, v8f c) {
    return __builtin_amdgcn_wmma_f32_16x16x32_bf16(
        false, a, false, b, (short)0, c, false, false);
}

// ---------------------------------------------------------------------------
// Fused preprocessing, one launch, grid (64, 32, 3), block (32, 8):
//   z == 0 : logits split  (2048 blocks x 1024 elems == 2,097,152 exactly)
//   z == 1 : W_in  transpose+split  -> Wt_in_h/l  [DIN][HID]
//   z == 2 : W_out transpose+split  -> Wt_out_h/l [DIN][HID]
// ---------------------------------------------------------------------------
__global__ __launch_bounds__(256) void preprocess_kernel(
    const float* __restrict__ logits,
    const float* __restrict__ W_in, const float* __restrict__ W_out,
    __bf16* __restrict__ Lh,  __bf16* __restrict__ Ll,
    __bf16* __restrict__ Wih, __bf16* __restrict__ Wil,
    __bf16* __restrict__ Woh, __bf16* __restrict__ Wol)
{
    __shared__ float tile[32][33];
    const int tx = threadIdx.x, ty = threadIdx.y;

    if (blockIdx.z == 0) {
        // elementwise hi/lo split of logits
        const int tid  = ty * 32 + tx;                       // 0..255
        const int base = (blockIdx.y * 64 + blockIdx.x) * 1024;
        #pragma unroll
        for (int i = 0; i < 4; ++i) {
            const int idx = base + tid + 256 * i;
            float v = logits[idx];
            __bf16 hi = (__bf16)v;
            Lh[idx] = hi;
            Ll[idx] = (__bf16)(v - (float)hi);
        }
        return;
    }

    const float* W  = (blockIdx.z == 1) ? W_in : W_out;
    __bf16*      Th = (blockIdx.z == 1) ? Wih  : Woh;
    __bf16*      Tl = (blockIdx.z == 1) ? Wil  : Wol;

    const int n0 = blockIdx.x * 32, k0 = blockIdx.y * 32;
    #pragma unroll
    for (int i = 0; i < 4; ++i)
        tile[ty + 8 * i][tx] = W[(size_t)(k0 + ty + 8 * i) * DIN + n0 + tx];
    __syncthreads();
    #pragma unroll
    for (int i = 0; i < 4; ++i) {
        float v = tile[tx][ty + 8 * i];
        __bf16 hi = (__bf16)v;
        size_t idx = (size_t)(n0 + ty + 8 * i) * HID + k0 + tx;
        Th[idx] = hi;
        Tl[idx] = (__bf16)(v - (float)hi);
    }
}

// ---------------------------------------------------------------------------
// Stage 1 GEMM (bf16x3): C[m,n] = clamp(sum_k L[m,k] W[k,n] + bias[n])
// grid.z selects {W_in,b_in -> IVt} or {W_out,b_out -> OVt}.
//   A: Lh/Ll [MTOT][HID] bf16 row-major
//   B: Wt_h/l [N=2048][HID] bf16 (transposed: row n, contiguous k)
//   Output written TRANSPOSED + split: outT_h/l [N=2048][MTOT] bf16
// Block 128 = 4 waves (2x2). Wave tile 32(M) x 64(N). K step 32.
// ---------------------------------------------------------------------------
__global__ __launch_bounds__(128) void gemm_gen_kernel(
    const __bf16* __restrict__ Ah_, const __bf16* __restrict__ Al_,
    const __bf16* __restrict__ Wih, const __bf16* __restrict__ Wil,
    const __bf16* __restrict__ Woh, const __bf16* __restrict__ Wol,
    const float*  __restrict__ b_in, const float* __restrict__ b_out,
    __bf16* __restrict__ IVth, __bf16* __restrict__ IVtl,
    __bf16* __restrict__ OVth, __bf16* __restrict__ OVtl)
{
    const int z = blockIdx.z;
    const __bf16* Bh_  = z ? Woh : Wih;
    const __bf16* Bl_  = z ? Wol : Wil;
    const float*  bias = z ? b_out : b_in;
    __bf16* outTh = z ? OVth : IVth;
    __bf16* outTl = z ? OVtl : IVtl;

    const int lane  = threadIdx.x & 31;
    const int wave  = threadIdx.x >> 5;
    const int waveM = wave >> 1, waveN = wave & 1;
    const int m0 = blockIdx.y * 64  + waveM * 32;
    const int n0 = blockIdx.x * 128 + waveN * 64;
    const int hh = lane >> 4, l16 = lane & 15;

    v8f acc[2][4];
    v8f zero;
    #pragma unroll
    for (int r = 0; r < 8; ++r) zero[r] = 0.0f;
    #pragma unroll
    for (int mi = 0; mi < 2; ++mi)
        #pragma unroll
        for (int nj = 0; nj < 4; ++nj) acc[mi][nj] = zero;

    // 32-bit element offsets (uniform base + VGPR offset -> saddr form)
    unsigned aoff[2], boff[4];
    #pragma unroll
    for (int mi = 0; mi < 2; ++mi) aoff[mi] = (unsigned)(m0 + mi * 16 + l16) * HID;
    #pragma unroll
    for (int nj = 0; nj < 4; ++nj) boff[nj] = (unsigned)(n0 + nj * 16 + l16) * HID;

    for (int k0 = 0; k0 < HID; k0 += 32) {
        const unsigned ka = (unsigned)(k0 + 8 * hh);    // A: chunks at ka, ka+16
        const unsigned kb = (unsigned)(k0 + 8 * hh);    // B: chunks at kb, kb+16
        v16bf a_h[2], a_l[2];
        #pragma unroll
        for (int mi = 0; mi < 2; ++mi) {
            a_h[mi] = frag_ab(*(const v4u*)(Ah_ + aoff[mi] + ka),
                              *(const v4u*)(Ah_ + aoff[mi] + ka + 16));
            a_l[mi] = frag_ab(*(const v4u*)(Al_ + aoff[mi] + ka),
                              *(const v4u*)(Al_ + aoff[mi] + ka + 16));
        }
        #pragma unroll
        for (int nj = 0; nj < 4; ++nj) {
            v16bf b_h = frag_ab(*(const v4u*)(Bh_ + boff[nj] + kb),
                                *(const v4u*)(Bh_ + boff[nj] + kb + 16));
            v16bf b_l = frag_ab(*(const v4u*)(Bl_ + boff[nj] + kb),
                                *(const v4u*)(Bl_ + boff[nj] + kb + 16));
            #pragma unroll
            for (int mi = 0; mi < 2; ++mi) {
                acc[mi][nj] = wmma_bf16(a_h[mi], b_h, acc[mi][nj]);
                acc[mi][nj] = wmma_bf16(a_h[mi], b_l, acc[mi][nj]);
                acc[mi][nj] = wmma_bf16(a_l[mi], b_h, acc[mi][nj]);
            }
        }
    }

    // Epilogue: bias + clamp, hi/lo split, transposed store (16B per frag).
    // C layout: VGPR r -> row m0+mi*16 + r + 8*hh, col n0+nj*16+l16.
    #pragma unroll
    for (int mi = 0; mi < 2; ++mi) {
        #pragma unroll
        for (int nj = 0; nj < 4; ++nj) {
            const int col = n0 + nj * 16 + l16;
            const float bv = bias[col];
            v8bf hi8, lo8;
            #pragma unroll
            for (int r = 0; r < 8; ++r) {
                float v = acc[mi][nj][r] + bv;
                v = fminf(fmaxf(v, -CLAMP_V), CLAMP_V);
                __bf16 h = (__bf16)v;
                hi8[r] = h;
                lo8[r] = (__bf16)(v - (float)h);
            }
            const unsigned base = (unsigned)col * MTOT + (unsigned)(m0 + mi * 16 + 8 * hh);
            *(v8bf*)(outTh + base) = hi8;
            *(v8bf*)(outTl + base) = lo8;
        }
    }
}

// ---------------------------------------------------------------------------
// Stage 2 GEMM (bf16x3): out[b,j,n] = Wp[b,j,n] + sum_i IV[b,i,j] OV[b,i,n]
//   A: IVt_h/l [DIN][MTOT]  (row j, k = b*128 + i contiguous)
//   B: OVt_h/l [DOUT][MTOT] (row n, k = b*128 + i contiguous)
// K = DEPTH = 128 -> 4 unrolled steps of 32.
// weight_params tile prefetched at entry to overlap HBM with the WMMA loop.
// ---------------------------------------------------------------------------
__global__ __launch_bounds__(128) void gemm_outer_kernel(
    const __bf16* __restrict__ IVth, const __bf16* __restrict__ IVtl,
    const __bf16* __restrict__ OVth, const __bf16* __restrict__ OVtl,
    const float* __restrict__ Wp, float* __restrict__ out)
{
    const int b    = blockIdx.z;
    const unsigned koff = (unsigned)(b * DEPTH);
    const float* wpb  = Wp  + (size_t)b * DIN * DOUT;
    float*       outb = out + (size_t)b * DIN * DOUT;

    const int lane  = threadIdx.x & 31;
    const int wave  = threadIdx.x >> 5;
    const int waveM = wave >> 1, waveN = wave & 1;
    const int j0 = blockIdx.y * 64  + waveM * 32;   // M: DIN index
    const int n0 = blockIdx.x * 128 + waveN * 64;   // N: DOUT index
    const int hh = lane >> 4, l16 = lane & 15;

    // Prefetch this wave's 32x64 weight_params tile (64 cachelines):
    // lane covers row j0+lane; two 128B lines per row.
    {
        const float* pf = wpb + (size_t)(j0 + lane) * DOUT + n0;
        __builtin_prefetch(pf, 0, 3);
        __builtin_prefetch(pf + 32, 0, 3);
    }

    v8f acc[2][4];
    v8f zero;
    #pragma unroll
    for (int r = 0; r < 8; ++r) zero[r] = 0.0f;
    #pragma unroll
    for (int mi = 0; mi < 2; ++mi)
        #pragma unroll
        for (int nj = 0; nj < 4; ++nj) acc[mi][nj] = zero;

    unsigned aoff[2], boff[4];
    #pragma unroll
    for (int mi = 0; mi < 2; ++mi)
        aoff[mi] = (unsigned)(j0 + mi * 16 + l16) * MTOT + koff;
    #pragma unroll
    for (int nj = 0; nj < 4; ++nj)
        boff[nj] = (unsigned)(n0 + nj * 16 + l16) * MTOT + koff;

    #pragma unroll
    for (int k0 = 0; k0 < DEPTH; k0 += 32) {
        const unsigned ka = (unsigned)(k0 + 8 * hh);
        const unsigned kb = (unsigned)(k0 + 8 * hh);
        v16bf a_h[2], a_l[2];
        #pragma unroll
        for (int mi = 0; mi < 2; ++mi) {
            a_h[mi] = frag_ab(*(const v4u*)(IVth + aoff[mi] + ka),
                              *(const v4u*)(IVth + aoff[mi] + ka + 16));
            a_l[mi] = frag_ab(*(const v4u*)(IVtl + aoff[mi] + ka),
                              *(const v4u*)(IVtl + aoff[mi] + ka + 16));
        }
        #pragma unroll
        for (int nj = 0; nj < 4; ++nj) {
            v16bf b_h = frag_ab(*(const v4u*)(OVth + boff[nj] + kb),
                                *(const v4u*)(OVth + boff[nj] + kb + 16));
            v16bf b_l = frag_ab(*(const v4u*)(OVtl + boff[nj] + kb),
                                *(const v4u*)(OVtl + boff[nj] + kb + 16));
            #pragma unroll
            for (int mi = 0; mi < 2; ++mi) {
                acc[mi][nj] = wmma_bf16(a_h[mi], b_h, acc[mi][nj]);
                acc[mi][nj] = wmma_bf16(a_h[mi], b_l, acc[mi][nj]);
                acc[mi][nj] = wmma_bf16(a_l[mi], b_h, acc[mi][nj]);
            }
        }
    }

    #pragma unroll
    for (int mi = 0; mi < 2; ++mi) {
        #pragma unroll
        for (int nj = 0; nj < 4; ++nj) {
            const int col = n0 + nj * 16 + l16;
            #pragma unroll
            for (int r = 0; r < 8; ++r) {
                const int row = j0 + mi * 16 + r + 8 * hh;
                const size_t idx = (size_t)row * DOUT + col;
                outb[idx] = wpb[idx] + acc[mi][nj][r];
            }
        }
    }
}

// ---------------------------------------------------------------------------
// Launch (3 launches total).
// Inputs: 0 weight_params [16,2048,2048] f32 | 1 logits [16,128,1024] f32
//         2 W_in [1024,2048] | 3 b_in [2048] | 4 W_out [1024,2048] | 5 b_out
// Workspace layout (bytes):
//   0  Lh 4MB | 4MB Ll | 8MB Wt_in_h | 12MB Wt_in_l | 16MB Wt_out_h
//   20MB Wt_out_l | 24MB IVt_h 8MB | 32MB IVt_l | 40MB OVt_h | 48MB OVt_l
//   total 56 MB
// ---------------------------------------------------------------------------
extern "C" void kernel_launch(void* const* d_in, const int* in_sizes, int n_in,
                              void* d_out, int out_size, void* d_ws, size_t ws_size,
                              hipStream_t stream) {
    const float* wp     = (const float*)d_in[0];
    const float* logits = (const float*)d_in[1];
    const float* W_in   = (const float*)d_in[2];
    const float* b_in   = (const float*)d_in[3];
    const float* W_out  = (const float*)d_in[4];
    const float* b_out  = (const float*)d_in[5];
    float* out = (float*)d_out;

    char* ws = (char*)d_ws;
    const size_t MB = 1024 * 1024;
    __bf16* Lh   = (__bf16*)(ws + 0 * MB);
    __bf16* Ll   = (__bf16*)(ws + 4 * MB);
    __bf16* Wih  = (__bf16*)(ws + 8 * MB);
    __bf16* Wil  = (__bf16*)(ws + 12 * MB);
    __bf16* Woh  = (__bf16*)(ws + 16 * MB);
    __bf16* Wol  = (__bf16*)(ws + 20 * MB);
    __bf16* IVth = (__bf16*)(ws + 24 * MB);
    __bf16* IVtl = (__bf16*)(ws + 32 * MB);
    __bf16* OVth = (__bf16*)(ws + 40 * MB);
    __bf16* OVtl = (__bf16*)(ws + 48 * MB);

    dim3 pblk(32, 8);
    dim3 pgrid(DIN / 32, HID / 32, 3);     // 64 x 32 x 3
    preprocess_kernel<<<pgrid, pblk, 0, stream>>>(
        logits, W_in, W_out, Lh, Ll, Wih, Wil, Woh, Wol);

    dim3 gblk(128, 1, 1);
    dim3 grid1(DIN / 128, MTOT / 64, 2);   // 16 x 32 x 2  (z: IV / OV)
    gemm_gen_kernel<<<grid1, gblk, 0, stream>>>(
        Lh, Ll, Wih, Wil, Woh, Wol, b_in, b_out, IVth, IVtl, OVth, OVtl);

    dim3 grid2(DOUT / 128, DIN / 64, NB);  // 16 x 32 x 16
    gemm_outer_kernel<<<grid2, gblk, 0, stream>>>(IVth, IVtl, OVth, OVtl, wp, out);
}
